// Attention_20667382628375
// MI455X (gfx1250) — compile-verified, matrix operand order checked by reading, not run on hardware
//
#include <hip/hip_runtime.h>
#include <hip/hip_bf16.h>
#include <math.h>

#define B_  2
#define N_  2048
#define L_  2048
#define E_  1024
#define H_  16
#define D_  4
#define C_  64
#define FC_ 256   // 4*C, per-head q/k dim

typedef __bf16 bf16;
typedef __attribute__((ext_vector_type(16))) __bf16 bf16x16;
typedef __attribute__((ext_vector_type(8)))  __bf16 bf16x8;
typedef __attribute__((ext_vector_type(8)))  float  f32x8;

// ---------------------------------------------------------------------------
// WMMA helpers. CDNA5 16-bit A (16x32) layout: lanes 0-15 hold M=lane with
// K = [0..7] (v0-3) and [16..23] (v4-7); lanes 16-31 hold same M with
// K = [8..15] and [24..31].  B (32x16) is the mirror image loaded from an
// [N,K] row-major ("column major B") array.  So a fragment is two contiguous
// 8-element (16B) chunks at k0+8*hi and k0+16+8*hi from row (lane&15).
// ---------------------------------------------------------------------------
static __device__ inline bf16x16 frag_ld(const bf16* base, int ld, int row0,
                                         int k0, int lane) {
  int r  = row0 + (lane & 15);
  int hi = lane >> 4;
  const bf16* p = base + (size_t)r * ld + k0 + 8 * hi;
  bf16x8 lo  = *(const bf16x8*)p;
  bf16x8 hi8 = *(const bf16x8*)(p + 16);
  bf16x16 f;
#pragma unroll
  for (int i = 0; i < 8; i++) { f[i] = lo[i]; f[8 + i] = hi8[i]; }
  return f;
}

static __device__ inline f32x8 wmma_bf16(bf16x16 a, bf16x16 b, f32x8 c) {
  return __builtin_amdgcn_wmma_f32_16x16x32_bf16(false, a, false, b,
                                                 (short)0, c, false, false);
}

// ---------------------------------------------------------------------------
// CDNA5 async global->LDS copy (ASYNCcnt tracked).  Per ISA 10.2 the LDS
// address is the low 32 bits of the flat address of a __shared__ object.
// ---------------------------------------------------------------------------
static __device__ inline unsigned lds_off32(const void* p) {
  return (unsigned)(uintptr_t)p;
}
static __device__ inline void async_ld_b128(unsigned lds, const bf16* g) {
  asm volatile("global_load_async_to_lds_b128 %0, %1, off"
               :: "v"(lds), "v"(g) : "memory");
}
static __device__ inline void wait_async0() {
  asm volatile("s_wait_asynccnt 0" ::: "memory");
}

// ---------------------------------------------------------------------------
// fp32 -> bf16 convert (weights)
// ---------------------------------------------------------------------------
__global__ void k_cvt(const float* __restrict__ s, bf16* __restrict__ d, size_t n) {
  size_t i = (size_t)blockIdx.x * 256 + threadIdx.x;
  if (i < n) d[i] = (bf16)s[i];
}

// ---------------------------------------------------------------------------
// LayerNorm over E=1024, one block per row, bf16 output
// ---------------------------------------------------------------------------
__global__ __launch_bounds__(256) void k_ln(const float* __restrict__ x,
                                            const float* __restrict__ g,
                                            const float* __restrict__ bta,
                                            bf16* __restrict__ o) {
  int row = blockIdx.x;
  const float* xr = x + (size_t)row * E_;
  float s = 0.f, s2 = 0.f;
  for (int i = threadIdx.x; i < E_; i += 256) { float v = xr[i]; s += v; s2 += v * v; }
#pragma unroll
  for (int off = 16; off >= 1; off >>= 1) { s += __shfl_xor(s, off); s2 += __shfl_xor(s2, off); }
  __shared__ float rs[8], rs2[8];
  __shared__ float smu, sinv;
  int w = threadIdx.x >> 5;
  if ((threadIdx.x & 31) == 0) { rs[w] = s; rs2[w] = s2; }
  __syncthreads();
  if (threadIdx.x == 0) {
    float a = 0.f, b2 = 0.f;
    for (int i = 0; i < 8; i++) { a += rs[i]; b2 += rs2[i]; }
    float m = a / E_;
    float var = b2 / E_ - m * m;
    smu = m; sinv = rsqrtf(var + 1e-5f);
  }
  __syncthreads();
  float mu = smu, inv = sinv;
  bf16* orow = o + (size_t)row * E_;
  for (int i = threadIdx.x; i < E_; i += 256)
    orow[i] = (bf16)((xr[i] - mu) * inv * g[i] + bta[i]);
}

// ---------------------------------------------------------------------------
// Generic bf16 WMMA GEMM: C[M,N] = act(A[M,K] @ W[N,K]^T + bias)
// Tiles: BM=128, BN=128, BK=32; double-buffered LDS filled by CDNA5 async
// global->LDS loads; one barrier per K-step.  8 waves in a 4x2 grid, each
// computing a 32x64 region (2x4 16x16x32 frags).
// act: 0=none, 1=exact GELU.  If Cb!=null store bf16 else fp32 (beta adds).
// Requires M%128==0, N%128==0, K%32==0 (true for all call sites here).
// ---------------------------------------------------------------------------
__global__ __launch_bounds__(256) void k_gemm(const bf16* __restrict__ A,
                                              const bf16* __restrict__ W,
                                              const float* __restrict__ bias,
                                              float* Cf, bf16* Cb,
                                              int M, int N, int K,
                                              int act, int beta) {
  __shared__ bf16 As[2][128 * 32];   // 2 x 8KB
  __shared__ bf16 Bs[2][128 * 32];   // 2 x 8KB
  int tid = threadIdx.x, lane = tid & 31, w = tid >> 5;
  int bm = blockIdx.y * 128, bn = blockIdx.x * 128;
  int wm0 = (w >> 1) * 32, wn0 = (w & 1) * 64;

  // per-thread staging slots: row = tid/2, col = (tid&1)*16 (two b128 each)
  int srow = tid >> 1, scol = (tid & 1) * 16;
  const bf16* aSrc = A + (size_t)(bm + srow) * K + scol;
  const bf16* bSrc = W + (size_t)(bn + srow) * K + scol;
  unsigned aDst0 = lds_off32(&As[0][srow * 32 + scol]);
  unsigned bDst0 = lds_off32(&Bs[0][srow * 32 + scol]);
  unsigned aDst1 = lds_off32(&As[1][srow * 32 + scol]);
  unsigned bDst1 = lds_off32(&Bs[1][srow * 32 + scol]);

  // prologue: stage k0 = 0 into buffer 0
  async_ld_b128(aDst0,     aSrc);
  async_ld_b128(aDst0 + 16, aSrc + 8);
  async_ld_b128(bDst0,     bSrc);
  async_ld_b128(bDst0 + 16, bSrc + 8);
  wait_async0();
  __syncthreads();

  f32x8 z = {0.f, 0.f, 0.f, 0.f, 0.f, 0.f, 0.f, 0.f};
  f32x8 acc[2][4];
#pragma unroll
  for (int mi = 0; mi < 2; mi++)
#pragma unroll
    for (int ni = 0; ni < 4; ni++) acc[mi][ni] = z;

  int nk = K / 32;
  int buf = 0;
  for (int kt = 0; kt < nk; kt++) {
    if (kt + 1 < nk) {   // async-stage next tile into the other buffer
      const bf16* aN = aSrc + (size_t)(kt + 1) * 32;
      const bf16* bN = bSrc + (size_t)(kt + 1) * 32;
      unsigned aD = buf ? aDst0 : aDst1;
      unsigned bD = buf ? bDst0 : bDst1;
      async_ld_b128(aD,      aN);
      async_ld_b128(aD + 16, aN + 8);
      async_ld_b128(bD,      bN);
      async_ld_b128(bD + 16, bN + 8);
    }
    // compute from current buffer
    bf16x16 af[2], bfr[4];
#pragma unroll
    for (int mi = 0; mi < 2; mi++) af[mi] = frag_ld(As[buf], 32, wm0 + mi * 16, 0, lane);
#pragma unroll
    for (int ni = 0; ni < 4; ni++) bfr[ni] = frag_ld(Bs[buf], 32, wn0 + ni * 16, 0, lane);
#pragma unroll
    for (int mi = 0; mi < 2; mi++)
#pragma unroll
      for (int ni = 0; ni < 4; ni++)
        acc[mi][ni] = wmma_bf16(af[mi], bfr[ni], acc[mi][ni]);

    if (kt + 1 < nk) {
      wait_async0();        // own async stores to LDS complete
      __syncthreads();      // everyone done reading old + writing new
      buf ^= 1;
    }
  }

#pragma unroll
  for (int mi = 0; mi < 2; mi++)
#pragma unroll
    for (int ni = 0; ni < 4; ni++) {
      int n  = bn + wn0 + ni * 16 + (lane & 15);
      int m0 = bm + wm0 + mi * 16 + 8 * (lane >> 4);
      float bv = bias ? bias[n] : 0.f;
#pragma unroll
      for (int i = 0; i < 8; i++) {
        float v = acc[mi][ni][i] + bv;
        if (act == 1) v = 0.5f * v * (1.f + erff(v * 0.70710678f));
        size_t idx = (size_t)(m0 + i) * N + n;
        if (Cb) Cb[idx] = (bf16)v;
        else { if (beta) v += Cf[idx]; Cf[idx] = v; }
      }
    }
}

// ---------------------------------------------------------------------------
// Position-encoder + q build:  q[b,h,n, 0:64)=x1*cos(xp1) |64:sin |128:x2*cos2
// xp1[e] = dot4(xp, p_w1[e]) + p_b1[e]   (ALPHA_Q = 1)
// ---------------------------------------------------------------------------
__global__ void k_build_q(const float* __restrict__ x1, const float* __restrict__ x2,
                          const float* __restrict__ xp,
                          const float* __restrict__ pw1, const float* __restrict__ pb1,
                          const float* __restrict__ pw2, const float* __restrict__ pb2,
                          bf16* __restrict__ q) {
  size_t gid = (size_t)blockIdx.x * 256 + threadIdx.x;
  if (gid >= (size_t)B_ * N_ * E_) return;
  int e = (int)(gid % E_);
  size_t bn = gid / E_;
  int b = (int)(bn / N_), n = (int)(bn % N_);
  const float* xpr = xp + bn * D_;
  float p0 = xpr[0], p1 = xpr[1], p2 = xpr[2], p3 = xpr[3];
  float a1 = p0 * pw1[e*4+0] + p1 * pw1[e*4+1] + p2 * pw1[e*4+2] + p3 * pw1[e*4+3] + pb1[e];
  float a2 = p0 * pw2[e*4+0] + p1 * pw2[e*4+1] + p2 * pw2[e*4+2] + p3 * pw2[e*4+3] + pb2[e];
  float v1 = x1[gid], v2 = x2[gid];
  int h = e >> 6, c = e & 63;
  bf16* dst = q + ((size_t)(b * H_ + h) * N_ + n) * FC_;
  dst[c]       = (bf16)(v1 * __cosf(a1));
  dst[c + 64]  = (bf16)(v1 * __sinf(a1));
  dst[c + 128] = (bf16)(v2 * __cosf(a2));
  dst[c + 192] = (bf16)(v2 * __sinf(a2));
}

// k build: no bias on yp projections; last two blocks are the raw pos code
__global__ void k_build_k(const float* __restrict__ y1, const float* __restrict__ yp,
                          const float* __restrict__ pw1, const float* __restrict__ pw2,
                          bf16* __restrict__ k) {
  size_t gid = (size_t)blockIdx.x * 256 + threadIdx.x;
  if (gid >= (size_t)B_ * L_ * E_) return;
  int e = (int)(gid % E_);
  size_t bl = gid / E_;
  int b = (int)(bl / L_), l = (int)(bl % L_);
  const float* ypr = yp + bl * D_;
  float p0 = ypr[0], p1 = ypr[1], p2 = ypr[2], p3 = ypr[3];
  float a1 = p0 * pw1[e*4+0] + p1 * pw1[e*4+1] + p2 * pw1[e*4+2] + p3 * pw1[e*4+3];
  float a2 = p0 * pw2[e*4+0] + p1 * pw2[e*4+1] + p2 * pw2[e*4+2] + p3 * pw2[e*4+3];
  float v1 = y1[gid];
  int h = e >> 6, c = e & 63;
  bf16* dst = k + ((size_t)(b * H_ + h) * L_ + l) * FC_;
  dst[c]       = (bf16)(v1 * __cosf(a1));
  dst[c + 64]  = (bf16)(v1 * __sinf(a1));
  dst[c + 128] = (bf16)__cosf(a2);
  dst[c + 192] = (bf16)__sinf(a2);
}

// v [B*L,E] f32 -> head-major transposed bf16 v_t [B,H,C,L]
__global__ void k_pack_vt(const float* __restrict__ vf, bf16* __restrict__ vt) {
  size_t i = (size_t)blockIdx.x * 256 + threadIdx.x;
  if (i >= (size_t)B_ * H_ * C_ * L_) return;
  int l = (int)(i % L_); size_t t = i / L_;
  int c = (int)(t % C_); t /= C_;
  int h = (int)(t % H_); int b = (int)(t / H_);
  vt[i] = (bf16)vf[((size_t)b * L_ + l) * E_ + h * C_ + c];
}

// yp [B,L,4] -> bf16 ypt [B,16,L] (rows 4..15 zero-padded for WMMA N-tile)
__global__ void k_pack_ypt(const float* __restrict__ yp, bf16* __restrict__ ypt) {
  size_t i = (size_t)blockIdx.x * 256 + threadIdx.x;
  if (i >= (size_t)B_ * 16 * L_) return;
  int l = (int)(i % L_); size_t t = i / L_;
  int r = (int)(t % 16); int b = (int)(t / 16);
  ypt[i] = (r < D_) ? (bf16)yp[((size_t)b * L_ + l) * D_ + r] : (bf16)0.f;
}

// ---------------------------------------------------------------------------
// Fused attention: one block = 16 query rows x full L for one (b,h).
// Phase 1: S = (q k^T)/16 into LDS via WMMA (q frags register-resident),
//          track row max.
// Phase 2: e = exp(s-max); t=sum e; u=sum e*mask; keep e*mask in LDS.
// Phase 3: attn = e*m/(u + 1e-8*t) streamed to d_out; then WMMA attn @ [V^T|yp]
// ---------------------------------------------------------------------------
__global__ __launch_bounds__(256) void k_attn(const bf16* __restrict__ qb,
                                              const bf16* __restrict__ kb,
                                              const bf16* __restrict__ vt,
                                              const bf16* __restrict__ ypt,
                                              const float* __restrict__ mask,
                                              float* __restrict__ attn,
                                              float* __restrict__ ov,
                                              float* __restrict__ op) {
  __shared__ float S[16 * L_];        // 128 KB score stripe (CDNA5: 320KB/WG)
  __shared__ bf16  qs[16 * FC_];      // 8 KB staged q rows
  __shared__ float wred[8 * 16];
  __shared__ float gmax[16];
  __shared__ float tred[16], ured[16], sinv[16];

  int tid = threadIdx.x, w = tid >> 5, lane = tid & 31;
  int n0 = blockIdx.x * 16, h = blockIdx.y, b = blockIdx.z;
  size_t bh = (size_t)b * H_ + h;

  {   // stage q: 16 rows x 256 bf16
    int row = tid >> 4, col = (tid & 15) * 16;
    const bf16* src = qb + (bh * N_ + n0) * FC_ + (size_t)row * FC_ + col;
    *(bf16x8*)&qs[row * FC_ + col]     = *(const bf16x8*)src;
    *(bf16x8*)&qs[row * FC_ + col + 8] = *(const bf16x8*)(src + 8);
  }
  if (tid < 16) { tred[tid] = 0.f; ured[tid] = 0.f; }
  __syncthreads();

  // hoist the 8 loop-invariant q fragments into registers
  bf16x16 qf[8];
#pragma unroll
  for (int kk = 0; kk < FC_ / 32; kk++) qf[kk] = frag_ld(qs, FC_, 0, kk * 32, lane);

  // ---- phase 1: scores ----
  float rmax[8];
#pragma unroll
  for (int i = 0; i < 8; i++) rmax[i] = -3.4e38f;
  const bf16* kbase = kb + bh * (size_t)L_ * FC_;
  f32x8 z = {0.f, 0.f, 0.f, 0.f, 0.f, 0.f, 0.f, 0.f};
  for (int lt = w; lt < L_ / 16; lt += 8) {
    int l0 = lt * 16;
    if (lt + 8 < L_ / 16)   // prefetch this wave's next k L-tile row
      __builtin_prefetch(kbase + (size_t)(l0 + 128 + (lane & 15)) * FC_, 0, 0);
    f32x8 acc = z;
#pragma unroll
    for (int kk = 0; kk < FC_ / 32; kk++) {
      bf16x16 bfr = frag_ld(kbase, FC_, l0, kk * 32, lane);
      acc = wmma_bf16(qf[kk], bfr, acc);
    }
    int col = l0 + (lane & 15);
    int rb  = 8 * (lane >> 4);
#pragma unroll
    for (int i = 0; i < 8; i++) {
      float v = acc[i] * 0.0625f;        // (4C)^-0.5 = 1/16
      S[(rb + i) * L_ + col] = v;
      rmax[i] = fmaxf(rmax[i], v);
    }
  }
#pragma unroll
  for (int i = 0; i < 8; i++) {
    float v = rmax[i];
    v = fmaxf(v, __shfl_xor(v, 1));
    v = fmaxf(v, __shfl_xor(v, 2));
    v = fmaxf(v, __shfl_xor(v, 4));
    v = fmaxf(v, __shfl_xor(v, 8));
    rmax[i] = v;
  }
  if (lane == 0)  for (int i = 0; i < 8; i++) wred[w * 16 + i]     = rmax[i];
  if (lane == 16) for (int i = 0; i < 8; i++) wred[w * 16 + 8 + i] = rmax[i];
  __syncthreads();
  if (tid < 16) {
    float g = -3.4e38f;
    for (int ww = 0; ww < 8; ww++) g = fmaxf(g, wred[ww * 16 + tid]);
    gmax[tid] = g;
  }
  __syncthreads();

  // ---- phase 2: exp + masked sums ----
  const float* mrow0 = mask + (size_t)n0 * L_;
  for (int r = 0; r < 16; r++) {
    float* Sr = &S[r * L_];
    const float* mr = mrow0 + (size_t)r * L_;
    float gm = gmax[r];
    float ts = 0.f, us = 0.f;
    for (int c = tid; c < L_; c += 256) {
      float e = __expf(Sr[c] - gm);
      float m = mr[c];
      ts += e; us += e * m;
      Sr[c] = e * m;
    }
#pragma unroll
    for (int off = 16; off >= 1; off >>= 1) { ts += __shfl_xor(ts, off); us += __shfl_xor(us, off); }
    if (lane == 0) { atomicAdd(&tred[r], ts); atomicAdd(&ured[r], us); }
  }
  __syncthreads();
  if (tid < 16) sinv[tid] = 1.f / (ured[tid] + 1e-8f * tred[tid]);
  __syncthreads();

  // ---- phase 3a: stream final attn out (the 512MB tensor, written once) ----
  float* arow = attn + (bh * N_ + n0) * (size_t)L_;
  for (int r = 0; r < 16; r++) {
    float inv = sinv[r];
    float* Sr = &S[r * L_];
    float* ar = arow + (size_t)r * L_;
    for (int c = tid; c < L_; c += 256) ar[c] = Sr[c] * inv;
  }

  // ---- phase 3b: attn @ [V^T (4 N-tiles) | yp (1 padded N-tile)] ----
  if (w < 5) {
    f32x8 acc = z;
    int rA = lane & 15, hiA = lane >> 4;
    float invA = sinv[rA];
    const bf16* vbase = vt + bh * (size_t)C_ * L_;
    const bf16* pbase = ypt + (size_t)b * 16 * L_;
    for (int kk = 0; kk < L_ / 32; kk++) {
      int k0 = kk * 32;
      const float* pA = &S[rA * L_ + k0 + 8 * hiA];
      bf16x16 af;
#pragma unroll
      for (int i = 0; i < 8; i++) {
        af[i]     = (bf16)(pA[i] * invA);
        af[8 + i] = (bf16)(pA[16 + i] * invA);
      }
      bf16x16 bfr = (w < 4) ? frag_ld(vbase, L_, w * 16, k0, lane)
                            : frag_ld(pbase, L_, 0,      k0, lane);
      acc = wmma_bf16(af, bfr, acc);
    }
    int n  = lane & 15;
    int m0 = 8 * (lane >> 4);
    if (w < 4) {
#pragma unroll
      for (int i = 0; i < 8; i++)
        ov[(bh * N_ + n0 + m0 + i) * (size_t)C_ + w * 16 + n] = acc[i];
    } else {
#pragma unroll
      for (int i = 0; i < 8; i++)
        op[(bh * N_ + n0 + m0 + i) * (size_t)16 + n] = acc[i];
    }
  }
}

// ov [B,H,N,C] -> A1 bf16 [B*N, E]   (BHNC -> BN(HC))
__global__ void k_pack_a1(const float* __restrict__ ov, bf16* __restrict__ A1) {
  size_t i = (size_t)blockIdx.x * 256 + threadIdx.x;
  if (i >= (size_t)B_ * N_ * E_) return;
  int e = (int)(i % E_); size_t bn = i / E_;
  int n = (int)(bn % N_); int b = (int)(bn / N_);
  int h = e >> 6, c = e & 63;
  A1[i] = (bf16)ov[((size_t)(b * H_ + h) * N_ + n) * C_ + c];
}

// (op - xp) -> A2 bf16 [B*N, H*D=64]
__global__ void k_pack_a2(const float* __restrict__ op, const float* __restrict__ xp,
                          bf16* __restrict__ A2) {
  size_t i = (size_t)blockIdx.x * 256 + threadIdx.x;
  if (i >= (size_t)B_ * N_ * 64) return;
  int e = (int)(i % 64); size_t bn = i / 64;
  int h = e >> 2, d = e & 3;
  A2[i] = (bf16)(op[((size_t)((bn / N_) * H_ + h) * N_ + (bn % N_)) * 16 + d]
                 - xp[bn * D_ + d]);
}

// ---------------------------------------------------------------------------
extern "C" void kernel_launch(void* const* d_in, const int* in_sizes, int n_in,
                              void* d_out, int out_size, void* d_ws, size_t ws_size,
                              hipStream_t stream) {
  (void)in_sizes; (void)n_in; (void)out_size; (void)ws_size;
  const float* x      = (const float*)d_in[0];
  const float* xp     = (const float*)d_in[1];
  const float* y      = (const float*)d_in[2];
  const float* yp     = (const float*)d_in[3];
  const float* amask  = (const float*)d_in[4];
  const float* p_w1   = (const float*)d_in[5];
  const float* p_b1   = (const float*)d_in[6];
  const float* p_w2   = (const float*)d_in[7];
  const float* p_b2   = (const float*)d_in[8];
  const float* q_ln_g = (const float*)d_in[9];
  const float* q_ln_b = (const float*)d_in[10];
  const float* qm1w1  = (const float*)d_in[11];
  const float* qm1b1  = (const float*)d_in[12];
  const float* qm1w2  = (const float*)d_in[13];
  const float* qm1b2  = (const float*)d_in[14];
  const float* qm2w1  = (const float*)d_in[15];
  const float* qm2b1  = (const float*)d_in[16];
  const float* qm2w2  = (const float*)d_in[17];
  const float* qm2b2  = (const float*)d_in[18];
  const float* k_ln_g = (const float*)d_in[19];
  const float* k_ln_b = (const float*)d_in[20];
  const float* km1w1  = (const float*)d_in[21];
  const float* km1b1  = (const float*)d_in[22];
  const float* km1w2  = (const float*)d_in[23];
  const float* km1b2  = (const float*)d_in[24];
  const float* v_ln_g = (const float*)d_in[25];
  const float* v_ln_b = (const float*)d_in[26];
  const float* v_w    = (const float*)d_in[27];
  const float* v_b    = (const float*)d_in[28];
  const float* p1_w   = (const float*)d_in[29];
  const float* p1_b   = (const float*)d_in[30];
  const float* p2_w   = (const float*)d_in[31];
  const float* p2_b   = (const float*)d_in[32];

  const size_t BN = (size_t)B_ * N_;   // 4096 rows (== B*L too)
  char* base = (char*)d_ws;
  size_t cur = 0;
  auto alloc = [&](size_t bytes) -> void* {
    void* p = base + cur;
    cur = (cur + bytes + 255) & ~(size_t)255;
    return p;
  };
  // bf16 weights
  bf16* w_q11 = (bf16*)alloc((size_t)2*E_*E_*2);
  bf16* w_q12 = (bf16*)alloc((size_t)2*E_*E_*2);
  bf16* w_q21 = (bf16*)alloc((size_t)2*E_*E_*2);
  bf16* w_q22 = (bf16*)alloc((size_t)2*E_*E_*2);
  bf16* w_k11 = (bf16*)alloc((size_t)2*E_*E_*2);
  bf16* w_k12 = (bf16*)alloc((size_t)2*E_*E_*2);
  bf16* w_v   = (bf16*)alloc((size_t)E_*E_*2);
  bf16* w_p1  = (bf16*)alloc((size_t)E_*E_*2);
  bf16* w_p2  = (bf16*)alloc((size_t)E_*H_*D_*2);
  // activations
  bf16*  xn   = (bf16*)alloc(BN*E_*2);
  bf16*  yn   = (bf16*)alloc(BN*E_*2);
  bf16*  vn   = (bf16*)alloc(BN*E_*2);
  bf16*  hbuf = (bf16*)alloc(BN*(size_t)2*E_*2);
  float* x1f  = (float*)alloc(BN*E_*4);
  float* x2f  = (float*)alloc(BN*E_*4);
  float* y1f  = (float*)alloc(BN*E_*4);
  float* vf   = (float*)alloc(BN*E_*4);
  bf16*  qbuf = (bf16*)alloc((size_t)B_*H_*N_*FC_*2);
  bf16*  kbuf = (bf16*)alloc((size_t)B_*H_*L_*FC_*2);
  bf16*  vt   = (bf16*)alloc((size_t)B_*H_*C_*L_*2);
  bf16*  ypt  = (bf16*)alloc((size_t)B_*16*L_*2);
  float* ovb  = (float*)alloc((size_t)B_*H_*N_*C_*4);
  float* opb  = (float*)alloc((size_t)B_*H_*N_*16*4);
  bf16*  A1   = (bf16*)alloc(BN*E_*2);
  bf16*  A2   = (bf16*)alloc(BN*64*2);

  float* out_main = (float*)d_out;                         // [B,N,E]
  float* out_attn = out_main + BN * E_;                    // [B,H,N,L]
  float* out_xp   = out_attn + (size_t)B_*H_*N_*L_;        // [B,N,D]

  auto cvt = [&](const float* s, bf16* d, size_t n) {
    k_cvt<<<dim3((unsigned)((n + 255) / 256)), 256, 0, stream>>>(s, d, n);
  };
  cvt(qm1w1, w_q11, (size_t)2*E_*E_);
  cvt(qm1w2, w_q12, (size_t)2*E_*E_);
  cvt(qm2w1, w_q21, (size_t)2*E_*E_);
  cvt(qm2w2, w_q22, (size_t)2*E_*E_);
  cvt(km1w1, w_k11, (size_t)2*E_*E_);
  cvt(km1w2, w_k12, (size_t)2*E_*E_);
  cvt(v_w,   w_v,   (size_t)E_*E_);
  cvt(p1_w,  w_p1,  (size_t)E_*E_);
  cvt(p2_w,  w_p2,  (size_t)E_*H_*D_);

  k_ln<<<dim3((unsigned)BN), 256, 0, stream>>>(x, q_ln_g, q_ln_b, xn);
  k_ln<<<dim3((unsigned)BN), 256, 0, stream>>>(y, k_ln_g, k_ln_b, yn);
  k_ln<<<dim3((unsigned)BN), 256, 0, stream>>>(y, v_ln_g, v_ln_b, vn);

  auto gemm = [&](const bf16* A, const bf16* W, const float* bias,
                  float* Cf, bf16* Cb, int M, int N, int K, int act, int beta) {
    k_gemm<<<dim3((unsigned)(N / 128), (unsigned)(M / 128)), 256, 0, stream>>>(
        A, W, bias, Cf, Cb, M, N, K, act, beta);
  };
  // Query MLPs
  gemm(xn,   w_q11, qm1b1, nullptr, hbuf, (int)BN, 2*E_, E_,   1, 0);
  gemm(hbuf, w_q12, qm1b2, x1f,  nullptr, (int)BN, E_,   2*E_, 0, 0);
  gemm(xn,   w_q21, qm2b1, nullptr, hbuf, (int)BN, 2*E_, E_,   1, 0);
  gemm(hbuf, w_q22, qm2b2, x2f,  nullptr, (int)BN, E_,   2*E_, 0, 0);
  // Key MLP
  gemm(yn,   w_k11, km1b1, nullptr, hbuf, (int)BN, 2*E_, E_,   1, 0);
  gemm(hbuf, w_k12, km1b2, y1f,  nullptr, (int)BN, E_,   2*E_, 0, 0);
  // V linear
  gemm(vn,   w_v,   v_b,   vf,   nullptr, (int)BN, E_,   E_,   0, 0);

  // q/k build with fused tiny position-encoder
  {
    size_t n = BN * E_;
    unsigned g = (unsigned)((n + 255) / 256);
    k_build_q<<<g, 256, 0, stream>>>(x1f, x2f, xp, p_w1, p_b1, p_w2, p_b2, qbuf);
    k_build_k<<<g, 256, 0, stream>>>(y1f, yp, p_w1, p_w2, kbuf);
  }
  {
    size_t n = (size_t)B_*H_*C_*L_;
    k_pack_vt<<<(unsigned)((n + 255) / 256), 256, 0, stream>>>(vf, vt);
  }
  {
    size_t n = (size_t)B_*16*L_;
    k_pack_ypt<<<(unsigned)((n + 255) / 256), 256, 0, stream>>>(yp, ypt);
  }

  // fused attention
  k_attn<<<dim3(N_/16, H_, B_), 256, 0, stream>>>(qbuf, kbuf, vt, ypt,
                                                  amask, out_attn, ovb, opb);

  // output projections
  {
    size_t n = BN * E_;
    k_pack_a1<<<(unsigned)((n + 255) / 256), 256, 0, stream>>>(ovb, A1);
    size_t n2 = BN * 64;
    k_pack_a2<<<(unsigned)((n2 + 255) / 256), 256, 0, stream>>>(opb, xp, A2);
  }
  gemm(A1, w_p1, p1_b, out_main, nullptr, (int)BN, E_, E_, 0, 0);
  gemm(A2, w_p2, p2_b, out_main, nullptr, (int)BN, E_, 64, 0, 1);

  // third output: xp passthrough
  hipMemcpyAsync(out_xp, xp, (size_t)B_*N_*D_*sizeof(float),
                 hipMemcpyDeviceToDevice, stream);
}
